// LSTMContext_25933012533726
// MI455X (gfx1250) — matrix operand order repaired and checked by reading, not compiled, not run on hardware
//
#include <hip/hip_runtime.h>
#include <hip/hip_bf16.h>
#include <cstdint>

// ---------------------------------------------------------------------------
// Problem constants (from reference)
// ---------------------------------------------------------------------------
#define TT      80
#define BB      256
#define EMBED   200
#define HIDV    512
#define NCLS    151
#define DDIM    4936        // HID + OBJ_DIM + EMBED + 128
#define N6      3072        // 6*HID
#define N5      2560        // 5*HID
#define KX      4960        // DDIM padded to multiple of 32
#define KE      224         // EMBED padded to multiple of 32
#define NPRED   160         // NCLS padded to multiple of 16
#define EROWS   152         // NCLS + 1 embedding rows
#define TB      (TT * BB)   // 20480

typedef __attribute__((ext_vector_type(16))) __bf16 v16bf;
typedef __attribute__((ext_vector_type(8)))  float  v8f;
typedef __attribute__((ext_vector_type(16))) short  v16s;
typedef __attribute__((ext_vector_type(8)))  short  v8s;

__device__ __forceinline__ v16bf cat16(v8s lo, v8s hi) {
    v16s s = __builtin_shufflevector(lo, hi, 0,1,2,3,4,5,6,7,8,9,10,11,12,13,14,15);
    return __builtin_bit_cast(v16bf, s);
}

__device__ __forceinline__ float sigmf(float x) {
    return 1.0f / (1.0f + __expf(-x));
}

// ---------------------------------------------------------------------------
// fp32 -> bf16 conversion with row/col zero-padding.
// ---------------------------------------------------------------------------
__global__ void cvt_pad_bf16(const float* __restrict__ src, int sstride, int scol0,
                             int srows, int scols,
                             __bf16* __restrict__ dst, int dcols, int rows) {
    size_t i = (size_t)blockIdx.x * 256 + threadIdx.x;
    size_t total = (size_t)rows * dcols;
    if (i >= total) return;
    int r = (int)(i / dcols);
    int c = (int)(i % dcols);
    float v = 0.0f;
    if (r < srows && c < scols) v = src[(size_t)r * sstride + scol0 + c];
    dst[i] = (__bf16)v;
}

__global__ void pad_bias_f32(const float* __restrict__ b, float* __restrict__ dst,
                             int n, int npad) {
    int i = blockIdx.x * 256 + threadIdx.x;
    if (i >= npad) return;
    dst[i] = (i < n) ? b[i] : 0.0f;
}

// h_bf = 0, c = 0, pe_bf[b][:] = emb_bf[0][:]
__global__ void init_state(float* __restrict__ cstate, __bf16* __restrict__ hbf,
                           __bf16* __restrict__ pebf, const __bf16* __restrict__ embbf) {
    int i = blockIdx.x * 256 + threadIdx.x;
    if (i < BB * HIDV) { cstate[i] = 0.0f; hbf[i] = (__bf16)0.0f; }
    if (i < BB * KE)   { pebf[i] = embbf[i % KE]; }
}

// ---------------------------------------------------------------------------
// Generic NT GEMM with bf16 WMMA, f32 accumulate:
//   Cout[M,N] = A[M,K] * B[N,K]^T  (+ Cin[M,N]) (+ bias[N])
// A row-major (lda), B row-major (ldb) -- K is contiguous in both.
// K must be a multiple of 32; M,N multiples of 16 (buffers padded).
// Block = 256 threads = 8 waves arranged 4(M) x 2(N); wave tile 64x64
// (4x4 grid of 16x16 WMMA tiles) -> block tile 256x128.
// Per K=32 iteration: 16 B128 loads feed 16 v_wmma (32 FLOP / cache byte).
// ---------------------------------------------------------------------------
__global__ void __launch_bounds__(256)
wmma_gemm_nt(const __bf16* __restrict__ A, int lda,
             const __bf16* __restrict__ Bm, int ldb,
             const float* __restrict__ Cin,
             const float* __restrict__ bias,
             float* __restrict__ Cout, int ldc,
             int M, int N, int K) {
    const int tid  = threadIdx.x;
    const int lane = tid & 31;
    const int wave = tid >> 5;       // 0..7
    const int wm   = wave >> 1;      // 0..3 (M direction)
    const int wn   = wave & 1;       // 0..1 (N direction)
    const int half = lane >> 4;      // 0/1
    const int l16  = lane & 15;

    const int m0 = blockIdx.y * 256 + wm * 64;
    const int n0 = blockIdx.x * 128 + wn * 64;

    int  mrow[4];  bool mok[4];
    int  ncol[4];  bool nok[4];
#pragma unroll
    for (int tm = 0; tm < 4; ++tm) {
        int r = m0 + tm * 16;
        mok[tm]  = (r < M);
        mrow[tm] = mok[tm] ? r : 0;      // clamp: loads stay in-bounds, store skipped
    }
#pragma unroll
    for (int tn = 0; tn < 4; ++tn) {
        int c = n0 + tn * 16;
        nok[tn]  = (c < N);
        ncol[tn] = nok[tn] ? c : 0;
    }

    v8f acc[4][4];
#pragma unroll
    for (int tm = 0; tm < 4; ++tm) {
#pragma unroll
        for (int tn = 0; tn < 4; ++tn) {
            v8f c = {};
            if (Cin != nullptr && mok[tm] && nok[tn]) {
                const float* p = Cin + (size_t)(mrow[tm] + half * 8) * ldc + ncol[tn] + l16;
#pragma unroll
                for (int r = 0; r < 8; ++r) c[r] = p[(size_t)r * ldc];
            }
            acc[tm][tn] = c;
        }
    }

    for (int kb = 0; kb < K; kb += 32) {
        if (kb + 64 <= K) {   // warm L2/WGP$ for the next K-block
            __builtin_prefetch(A  + (size_t)(mrow[0] + l16) * lda + kb + 32, 0, 1);
            __builtin_prefetch(Bm + (size_t)(ncol[0] + l16) * ldb + kb + 32, 0, 1);
        }
        v16bf af[4], bfr[4];
#pragma unroll
        for (int tm = 0; tm < 4; ++tm) {
            // A 16x32 bf16 fragment: lane holds row (l&15);
            // elems 0..7  <- K = kb + half*8      + 0..7
            // elems 8..15 <- K = kb + 16 + half*8 + 0..7
            const __bf16* pa = A + (size_t)(mrow[tm] + l16) * lda + kb + half * 8;
            v8s lo = *(const v8s*)(pa);
            v8s hi = *(const v8s*)(pa + 16);
            af[tm] = cat16(lo, hi);
        }
#pragma unroll
        for (int tn = 0; tn < 4; ++tn) {
            // B 32x16 bf16 fragment: lane holds col (l&15);
            // elems 0..7 <- K = kb + half*8 + 0..7; elems 8..15 <- K = kb + 16 + half*8 + 0..7
            const __bf16* pb = Bm + (size_t)(ncol[tn] + l16) * ldb + kb + half * 8;
            v8s lo = *(const v8s*)(pb);
            v8s hi = *(const v8s*)(pb + 16);
            bfr[tn] = cat16(lo, hi);
        }
#pragma unroll
        for (int tm = 0; tm < 4; ++tm) {
#pragma unroll
            for (int tn = 0; tn < 4; ++tn) {
                acc[tm][tn] = __builtin_amdgcn_wmma_f32_16x16x32_bf16(
                    false, af[tm], false, bfr[tn], (short)0, acc[tm][tn], false, false);
            }
        }
    }

#pragma unroll
    for (int tm = 0; tm < 4; ++tm) {
#pragma unroll
        for (int tn = 0; tn < 4; ++tn) {
            if (!mok[tm] || !nok[tn]) continue;
            float bv = (bias != nullptr) ? bias[ncol[tn] + l16] : 0.0f;
            float* p = Cout + (size_t)(mrow[tm] + half * 8) * ldc + ncol[tn] + l16;
#pragma unroll
            for (int r = 0; r < 8; ++r) p[(size_t)r * ldc] = acc[tm][tn][r] + bv;
        }
    }
}

// ---------------------------------------------------------------------------
// Gate / cell / highway element-wise math.  g: [BB, 3072] pre-activations.
// ---------------------------------------------------------------------------
__global__ void cell_kernel(const float* __restrict__ g, float* __restrict__ cstate,
                            __bf16* __restrict__ hbf) {
    int idx = blockIdx.x * 256 + threadIdx.x;
    if (idx >= BB * HIDV) return;
    int b = idx >> 9;          // /512
    int j = idx & (HIDV - 1);
    const float* gb = g + (size_t)b * N6;
    float i_g = sigmf(gb[0 * HIDV + j]);
    float f_g = sigmf(gb[1 * HIDV + j]);
    float m_i = tanhf(gb[2 * HIDV + j]);
    float o_g = sigmf(gb[3 * HIDV + j]);
    float hw  = sigmf(gb[4 * HIDV + j]);
    float p5  =       gb[5 * HIDV + j];
    float cn  = i_g * m_i + f_g * cstate[idx];
    float out = o_g * tanhf(cn);
    float hn  = hw * out + (1.0f - hw) * p5;
    cstate[idx] = cn;
    hbf[idx]    = (__bf16)hn;
}

// ---------------------------------------------------------------------------
// Per-step: copy pred row into allD, argmax over classes 1..150,
// gather next prev_embed row (embed[best+1]) in bf16.
// One block per batch element.
// ---------------------------------------------------------------------------
__global__ void argmax_kernel(const float* __restrict__ pred, float* __restrict__ allD,
                              int* __restrict__ allBest, __bf16* __restrict__ pebf,
                              const __bf16* __restrict__ embbf, int t) {
    int b   = blockIdx.x;
    int tid = threadIdx.x;
    const float* pb = pred + (size_t)b * NPRED;
    if (tid < NCLS) allD[((size_t)t * BB + b) * NCLS + tid] = pb[tid];
    __shared__ int sbest;
    if (tid == 0) {
        float mx = pb[1]; int bi = 1;
        for (int n = 2; n < NCLS; ++n) {
            float v = pb[n];
            if (v > mx) { mx = v; bi = n; }
        }
        sbest = bi;
        allBest[t * BB + b] = bi;
    }
    __syncthreads();
    int row = sbest + 1;     // pe_new = embed[best + 1]
    if (tid < KE) pebf[(size_t)b * KE + tid] = embbf[(size_t)row * KE + tid];
}

// ---------------------------------------------------------------------------
// Final packed gather: out = [dists (Np x 151) | commits (Np)]
// ---------------------------------------------------------------------------
__global__ void gather_kernel(const float* __restrict__ allD, const int* __restrict__ allBest,
                              const int* __restrict__ gidx, float* __restrict__ out,
                              int Np) {
    size_t i = (size_t)blockIdx.x * 256 + threadIdx.x;
    size_t total = (size_t)Np * (NCLS + 1);
    if (i >= total) return;
    int p = (int)(i / (NCLS + 1));
    int c = (int)(i % (NCLS + 1));
    int src = gidx[p];
    if (c < NCLS) out[(size_t)p * NCLS + c] = allD[(size_t)src * NCLS + c];
    else          out[(size_t)Np * NCLS + p] = (float)allBest[src];
}

// ---------------------------------------------------------------------------
// Host launcher
// ---------------------------------------------------------------------------
extern "C" void kernel_launch(void* const* d_in, const int* in_sizes, int n_in,
                              void* d_out, int out_size, void* d_ws, size_t ws_size,
                              hipStream_t stream) {
    const float* seq     = (const float*)d_in[0];   // [T,B,D]
    const float* embed   = (const float*)d_in[1];   // [152,200]
    const float* W_in    = (const float*)d_in[2];   // [3072, 5136]
    const float* b_in    = (const float*)d_in[3];   // [3072]
    const float* W_state = (const float*)d_in[4];   // [2560, 512]
    const float* b_state = (const float*)d_in[5];   // [2560]
    const float* W_out   = (const float*)d_in[6];   // [151, 512]
    const float* b_out   = (const float*)d_in[7];   // [151]
    const int*   gidx    = (const int*)d_in[8];
    const int Np = in_sizes[8];

    // ---- carve workspace ----
    char* w = (char*)d_ws;
    size_t off = 0;
    auto alloc = [&](size_t bytes) -> char* {
        char* p = w + off;
        off += (bytes + 255) & ~(size_t)255;
        return p;
    };
    __bf16* seq_bf   = (__bf16*)alloc((size_t)TB * KX * 2);      // padded sequence
    __bf16* winx_bf  = (__bf16*)alloc((size_t)N6 * KX * 2);      // W_in[:, :D]
    __bf16* wine_bf  = (__bf16*)alloc((size_t)N6 * KE * 2);      // W_in[:, D:]
    __bf16* wst_bf   = (__bf16*)alloc((size_t)N5 * HIDV * 2);    // W_state
    __bf16* wout_bf  = (__bf16*)alloc((size_t)NPRED * HIDV * 2); // W_out padded rows
    __bf16* emb_bf   = (__bf16*)alloc((size_t)EROWS * KE * 2);   // embed padded cols
    float*  bout_pad = (float*)alloc((size_t)NPRED * 4);
    float*  xproj    = (float*)alloc((size_t)TB * N6 * 4);       // precomputed x-part + b_in
    float*  gbuf     = (float*)alloc((size_t)BB * N6 * 4);
    float*  cstate   = (float*)alloc((size_t)BB * HIDV * 4);
    __bf16* h_bf     = (__bf16*)alloc((size_t)BB * HIDV * 2);
    __bf16* pe_bf    = (__bf16*)alloc((size_t)BB * KE * 2);
    float*  predbuf  = (float*)alloc((size_t)BB * NPRED * 4);
    float*  allD     = (float*)alloc((size_t)TB * NCLS * 4);
    int*    allBest  = (int*)alloc((size_t)TB * 4);

    auto blocks1d = [](size_t n) -> unsigned { return (unsigned)((n + 255) / 256); };

    // ---- one-time (per launch) conversions ----
    cvt_pad_bf16<<<blocks1d((size_t)TB * KX), 256, 0, stream>>>(
        seq, DDIM, 0, TB, DDIM, seq_bf, KX, TB);
    cvt_pad_bf16<<<blocks1d((size_t)N6 * KX), 256, 0, stream>>>(
        W_in, DDIM + EMBED, 0, N6, DDIM, winx_bf, KX, N6);
    cvt_pad_bf16<<<blocks1d((size_t)N6 * KE), 256, 0, stream>>>(
        W_in, DDIM + EMBED, DDIM, N6, EMBED, wine_bf, KE, N6);
    cvt_pad_bf16<<<blocks1d((size_t)N5 * HIDV), 256, 0, stream>>>(
        W_state, HIDV, 0, N5, HIDV, wst_bf, HIDV, N5);
    cvt_pad_bf16<<<blocks1d((size_t)NPRED * HIDV), 256, 0, stream>>>(
        W_out, HIDV, 0, NCLS, HIDV, wout_bf, HIDV, NPRED);
    cvt_pad_bf16<<<blocks1d((size_t)EROWS * KE), 256, 0, stream>>>(
        embed, EMBED, 0, EROWS, EMBED, emb_bf, KE, EROWS);
    pad_bias_f32<<<1, 256, 0, stream>>>(b_out, bout_pad, NCLS, NPRED);
    init_state<<<blocks1d((size_t)BB * HIDV), 256, 0, stream>>>(cstate, h_bf, pe_bf, emb_bf);

    // ---- phase 1: big precomputable GEMM  xproj = seq @ W_in_x^T + b_in ----
    {
        dim3 grid(N6 / 128, TB / 256);   // 24 x 80
        wmma_gemm_nt<<<grid, 256, 0, stream>>>(
            seq_bf, KX, winx_bf, KX, nullptr, b_in, xproj, N6, TB, N6, KX);
    }

    // ---- phase 2: 80 recurrent steps ----
    for (int t = 0; t < TT; ++t) {
        // g = xproj[t] + pe @ W_in_e^T            (all 3072 outputs)
        {
            dim3 grid(N6 / 128, (BB + 255) / 256);   // 24 x 1
            wmma_gemm_nt<<<grid, 256, 0, stream>>>(
                pe_bf, KE, wine_bf, KE,
                xproj + (size_t)t * BB * N6, nullptr, gbuf, N6, BB, N6, KE);
        }
        // g[:, :2560] += h @ W_state^T + b_state
        {
            dim3 grid(N5 / 128, (BB + 255) / 256);   // 20 x 1
            wmma_gemm_nt<<<grid, 256, 0, stream>>>(
                h_bf, HIDV, wst_bf, HIDV, gbuf, b_state, gbuf, N6, BB, N5, HIDV);
        }
        // gates / cell / highway
        cell_kernel<<<blocks1d((size_t)BB * HIDV), 256, 0, stream>>>(gbuf, cstate, h_bf);
        // pred = h_new @ W_out^T + b_out   (padded to 160 cols)
        {
            dim3 grid((NPRED + 127) / 128, (BB + 255) / 256);  // 2 x 1
            wmma_gemm_nt<<<grid, 256, 0, stream>>>(
                h_bf, HIDV, wout_bf, HIDV, nullptr, bout_pad, predbuf, NPRED,
                BB, NPRED, HIDV);
        }
        // argmax + store dists + fetch next prev_embed
        argmax_kernel<<<BB, 256, 0, stream>>>(predbuf, allD, allBest, pe_bf, emb_bf, t);
    }

    // ---- final packed gather into d_out ----
    gather_kernel<<<blocks1d((size_t)Np * (NCLS + 1)), 256, 0, stream>>>(
        allD, allBest, gidx, (float*)d_out, Np);
}
